// AttentionMixerModule_76295799046721
// MI455X (gfx1250) — matrix operand order repaired and checked by reading, not compiled, not run on hardware
//
#include <hip/hip_runtime.h>
#include <stddef.h>
#include <stdint.h>

// ---------------------------------------------------------------------------
// Problem constants (from the reference): B=2, S=2048, D=1024, H=16, HD=64,
// WINDOW=256.  Effective mask: query i attends to all j <= i+256.
// ---------------------------------------------------------------------------
#define SEQ    2048
#define DMODEL 1024
#define NHEAD  16
#define HDIM   64
#define NTOK   4096          // B * S
#define WIN    256

typedef unsigned short u16;
typedef unsigned int   u32;

typedef __attribute__((ext_vector_type(16))) __bf16 bf16x16;
typedef __attribute__((ext_vector_type(8)))  float  f32x8;
typedef int v4i __attribute__((vector_size(16)));   // matches builtin proto

union Frag { uint4 u[2]; bf16x16 v; };

static __device__ __forceinline__ int imin(int a, int b) { return a < b ? a : b; }

// round-to-nearest-even f32 -> bf16 (as raw u16)
static __device__ __forceinline__ u16 f2bf(float f) {
    u32 u = __float_as_uint(f);
    u32 r = u + 0x7fffu + ((u >> 16) & 1u);
    return (u16)(r >> 16);
}

// ---------------------------------------------------------------------------
// CDNA5 async global->LDS copy (ASYNCcnt path), with portable fallback.
// ---------------------------------------------------------------------------
#if defined(__has_builtin)
#if __has_builtin(__builtin_amdgcn_global_load_async_to_lds_b128) && \
    __has_builtin(__builtin_amdgcn_s_wait_asynccnt)
#define USE_ASYNC 1
#endif
#endif
#ifndef USE_ASYNC
#define USE_ASYNC 0
#endif

static __device__ __forceinline__ void copy16_g2lds(const u16* g, u16* l) {
#if USE_ASYNC
    // flat->AS1 is the same 64-bit address; flat shared truncated to 32 bits
    // is the LDS offset (ISA 10.2).
    __builtin_amdgcn_global_load_async_to_lds_b128(
        (__attribute__((address_space(1))) v4i*)(uintptr_t)g,
        (__attribute__((address_space(3))) v4i*)(u32)(uintptr_t)l,
        0, 0);
#else
    *(uint4*)l = *(const uint4*)g;
#endif
}

static __device__ __forceinline__ void async_wait0() {
#if USE_ASYNC
    __builtin_amdgcn_s_wait_asynccnt(0);
#endif
}

// ---------------------------------------------------------------------------
// Kernel 1: vectorized f32 -> bf16 conversion (4 elements / thread)
// ---------------------------------------------------------------------------
__global__ __launch_bounds__(256)
void cvt_f32_bf16_x4(const float* __restrict__ src, u16* __restrict__ dst, int n4) {
    int i = blockIdx.x * blockDim.x + threadIdx.x;
    if (i < n4) {
        float4 f = ((const float4*)src)[i];
        uint2 o;
        o.x = (u32)f2bf(f.x) | ((u32)f2bf(f.y) << 16);
        o.y = (u32)f2bf(f.z) | ((u32)f2bf(f.w) << 16);
        ((uint2*)dst)[i] = o;
    }
}

// ---------------------------------------------------------------------------
// Kernel 2: NT GEMM  C[t, n] = sum_k A[t, k] * W[n, k] + bias[n]
// A: [M, K] bf16 row-major; W: [N, K] bf16 row-major.
// Wave computes a 64x64 tile (16 accumulators); block = 4 waves = 128x128.
// Per-lane fragments use k = 16*half + s for both A and B, so loads are contiguous.
// ---------------------------------------------------------------------------
template <bool OUTF32>
__global__ __launch_bounds__(128)
void gemm_bf16_nt(const u16* __restrict__ A, const u16* __restrict__ W,
                  const float* __restrict__ bias, void* __restrict__ Cout,
                  int N, int K) {
    const int lane = threadIdx.x & 31;
    const int wave = threadIdx.x >> 5;
    const int m16  = lane & 15;
    const int half = lane >> 4;
    const int tok0  = blockIdx.y * 128 + (wave >> 1) * 64;
    const int feat0 = blockIdx.x * 128 + (wave & 1) * 64;

    f32x8 acc[4][4] = {};
    float bv[4];
#pragma unroll
    for (int c = 0; c < 4; ++c) bv[c] = bias[feat0 + c * 16 + m16];

    const u16* aRow[4];
    const u16* bRow[4];
#pragma unroll
    for (int r = 0; r < 4; ++r) aRow[r] = A + (size_t)(tok0 + r * 16 + m16) * K;
#pragma unroll
    for (int c = 0; c < 4; ++c) bRow[c] = W + (size_t)(feat0 + c * 16 + m16) * K;

    for (int k0 = 0; k0 < K; k0 += 32) {
        Frag af[4], bf[4];
#pragma unroll
        for (int r = 0; r < 4; ++r) {
            const u16* p = aRow[r] + k0;
            af[r].u[0] = *(const uint4*)(p + half * 16);       // K = half*16 .. +8
            af[r].u[1] = *(const uint4*)(p + half * 16 + 8);   // K = half*16+8 .. +8
        }
#pragma unroll
        for (int c = 0; c < 4; ++c) {
            const u16* p = bRow[c] + k0 + 16 * half;           // K = 16*half .. +16
            bf[c].u[0] = *(const uint4*)(p);
            bf[c].u[1] = *(const uint4*)(p + 8);
        }
        if (k0 + 32 < K) {                                     // global_prefetch_b8
            __builtin_prefetch(aRow[0] + k0 + 32, 0, 1);
            __builtin_prefetch(bRow[0] + k0 + 32, 0, 1);
        }
#pragma unroll
        for (int r = 0; r < 4; ++r)
#pragma unroll
            for (int c = 0; c < 4; ++c)
                acc[r][c] = __builtin_amdgcn_wmma_f32_16x16x32_bf16(
                    false, af[r].v, false, bf[c].v, (short)0, acc[r][c], false, false);
    }

#pragma unroll
    for (int r = 0; r < 4; ++r)
#pragma unroll
        for (int c = 0; c < 4; ++c)
#pragma unroll
            for (int v = 0; v < 8; ++v) {
                int tok  = tok0 + r * 16 + v + 8 * half;       // C layout: row = v + 8*half
                int feat = feat0 + c * 16 + m16;               // col = lane&15
                float val = acc[r][c][v] + bv[c];
                if (OUTF32) ((float*)Cout)[(size_t)tok * N + feat] = val;
                else        ((u16*)Cout)[(size_t)tok * N + feat]  = f2bf(val);
            }
}

// ---------------------------------------------------------------------------
// Kernel 3: flash attention, head_dim=64, keys j <= i+256 allowed.
// Block = 4 waves over one 64-query tile; wave w owns 16 queries.
// Block-cooperative, double-buffered staging:
//   K tile (32 keys x 64 d) -> LDS via GLOBAL_LOAD_ASYNC_TO_LDS_B128
//     (ASYNCcnt, settled with s_wait_asynccnt 0 before the barrier),
//   V tile transposed into LDS via packed 32-bit stores (loads issued before
//   compute, packing stores after -> software pipelined).
// Uniform trip count across waves; out-of-range / future keys are handled by
// the -1e9 mask (exp -> 0), so clamped loads of garbage rows are harmless.
// Per 32-key step: 4 WMMA for Q*K^T, online softmax (shfl_xor row
// reductions), P re-shaped through per-wave LDS, 4 WMMA for P*V.
// qkv: [NTOK, 3072] bf16 (q | k | v per token).  attn: [NTOK, 1024] bf16.
// ---------------------------------------------------------------------------
__global__ __launch_bounds__(128)
void flash_attn(const u16* __restrict__ qkv, u16* __restrict__ attn) {
    __shared__ __align__(16) u16 Ksh[2][32][64];    //  8 KB, K tiles, row-major
    __shared__ __align__(16) u16 Vt[2][64][32];     //  8 KB, V^T tiles
    __shared__ __align__(16) u16 Pbuf[4][16][32];   //  4 KB, per-wave P staging

    const int tid  = threadIdx.x;
    const int lane = tid & 31;
    const int w    = tid >> 5;
    const int m16  = lane & 15;
    const int half = lane >> 4;
    const int b    = blockIdx.x >> 4;
    const int h    = blockIdx.x & 15;
    const int qblk = blockIdx.y * 64;
    const int q0   = qblk + w * 16;

    const size_t RS = 3 * DMODEL;                    // token row stride
    const u16* base = qkv + (size_t)b * SEQ * RS;

    // ---- Q fragments (A layout), loaded once: 16 queries x 64 dims ----
    Frag aq[2];
#pragma unroll
    for (int dc = 0; dc < 2; ++dc) {
        const u16* qp = base + (size_t)(q0 + m16) * RS + h * HDIM + dc * 32;
        aq[dc].u[0] = *(const uint4*)(qp + half * 16);
        aq[dc].u[1] = *(const uint4*)(qp + half * 16 + 8);
    }

    f32x8 o[4] = {};
    float mrow[8], lrow[8];
#pragma unroll
    for (int v = 0; v < 8; ++v) { mrow[v] = -1e30f; lrow[v] = 0.0f; }

    // uniform across the block: covers max query row qblk+63
    const int nsteps = (imin(SEQ, qblk + 64 + WIN) + 31) >> 5;

    // ---- cooperative staging helpers ----
    // K tile: 256 chunks of 16B; thread handles chunks tid and tid+128
    auto stageK = [&](int j0, int buf) {
#pragma unroll
        for (int i2 = 0; i2 < 2; ++i2) {
            int c   = tid + i2 * 128;
            int row = c >> 3, sub = c & 7;
            int key = imin(j0 + row, SEQ - 1);
            const u16* g = base + (size_t)key * RS + DMODEL + h * HDIM + sub * 8;
            copy16_g2lds(g, &Ksh[buf][row][sub * 8]);
        }
    };
    // V tile: thread owns key pair (2*vkp, 2*vkp+1) x 8 dims [vseg*8, vseg*8+8)
    const int vkp  = tid & 15;
    const int vseg = tid >> 4;
    uint4 vldA, vldB;
    auto loadV = [&](int j0) {
        int ra = imin(j0 + 2 * vkp,     SEQ - 1);
        int rb = imin(j0 + 2 * vkp + 1, SEQ - 1);
        vldA = *(const uint4*)(base + (size_t)ra * RS + 2 * DMODEL + h * HDIM + vseg * 8);
        vldB = *(const uint4*)(base + (size_t)rb * RS + 2 * DMODEL + h * HDIM + vseg * 8);
    };
    auto storeV = [&](int buf) {
        const u32* ua = (const u32*)&vldA;
        const u32* ub = (const u32*)&vldB;
#pragma unroll
        for (int q = 0; q < 4; ++q) {
            int d = vseg * 8 + 2 * q;
            *(u32*)&Vt[buf][d][2 * vkp]     = (ua[q] & 0xffffu) | (ub[q] << 16);
            *(u32*)&Vt[buf][d + 1][2 * vkp] = (ua[q] >> 16)     | (ub[q] & 0xffff0000u);
        }
    };

    // ---- prologue: stage tile 0 ----
    stageK(0, 0);
    loadV(0);
    storeV(0);
    async_wait0();
    __syncthreads();

    for (int i = 0; i < nsteps; ++i) {
        const int j0   = i << 5;
        const int cur  = i & 1;
        const bool more = (i + 1) < nsteps;

        if (more) { stageK(j0 + 32, cur ^ 1); loadV(j0 + 32); }

        // ---- S = Q * K^T from the LDS K tile ----
        f32x8 s[2];
#pragma unroll
        for (int kt = 0; kt < 2; ++kt) {
            const u16* kr = &Ksh[cur][kt * 16 + m16][16 * half];
            Frag kb0, kb1;
            kb0.u[0] = *(const uint4*)(kr);       kb0.u[1] = *(const uint4*)(kr + 8);
            kb1.u[0] = *(const uint4*)(kr + 32);  kb1.u[1] = *(const uint4*)(kr + 40);
            f32x8 t = {};
            t = __builtin_amdgcn_wmma_f32_16x16x32_bf16(false, aq[0].v, false, kb0.v, (short)0, t, false, false);
            t = __builtin_amdgcn_wmma_f32_16x16x32_bf16(false, aq[1].v, false, kb1.v, (short)0, t, false, false);
            s[kt] = t;
        }

        // ---- online softmax + write P (bf16) to this wave's LDS slice ----
#pragma unroll
        for (int v = 0; v < 8; ++v) {
            const int irel = v + 8 * half;                 // query row in tile
            const int jrel = j0 - q0 + m16;                // key col (sub-tile 0)
            float s1 = s[0][v] * 0.125f;                   // scale = HDIM^-0.5
            float s2 = s[1][v] * 0.125f;
            if (jrel      > irel + WIN) s1 += -1e9f;       // mask (matches reference add)
            if (jrel + 16 > irel + WIN) s2 += -1e9f;

            float tm = fmaxf(s1, s2);
            tm = fmaxf(tm, __shfl_xor(tm, 1, 16));
            tm = fmaxf(tm, __shfl_xor(tm, 2, 16));
            tm = fmaxf(tm, __shfl_xor(tm, 4, 16));
            tm = fmaxf(tm, __shfl_xor(tm, 8, 16));
            float mn  = fmaxf(mrow[v], tm);
            float fct = __expf(mrow[v] - mn);
            float p1  = __expf(s1 - mn);
            float p2  = __expf(s2 - mn);
            float rs  = p1 + p2;
            rs += __shfl_xor(rs, 1, 16);
            rs += __shfl_xor(rs, 2, 16);
            rs += __shfl_xor(rs, 4, 16);
            rs += __shfl_xor(rs, 8, 16);
            lrow[v] = lrow[v] * fct + rs;
            mrow[v] = mn;
            Pbuf[w][irel][m16]      = f2bf(p1);
            Pbuf[w][irel][16 + m16] = f2bf(p2);
#pragma unroll
            for (int c = 0; c < 4; ++c) o[c][v] *= fct;
        }

        // ---- O += P * V : P as A fragment, V^T rows as B fragments ----
        Frag pa;
        pa.u[0] = *(const uint4*)&Pbuf[w][m16][half * 16];
        pa.u[1] = *(const uint4*)&Pbuf[w][m16][half * 16 + 8];
#pragma unroll
        for (int c = 0; c < 4; ++c) {
            Frag vb;
            vb.u[0] = *(const uint4*)&Vt[cur][c * 16 + m16][16 * half];
            vb.u[1] = *(const uint4*)&Vt[cur][c * 16 + m16][16 * half + 8];
            o[c] = __builtin_amdgcn_wmma_f32_16x16x32_bf16(
                false, pa.v, false, vb.v, (short)0, o[c], false, false);
        }

        // ---- finish pipelined staging of the next tile ----
        if (more) storeV(cur ^ 1);
        async_wait0();
        __syncthreads();
    }

    // ---- normalize and store merged-head output (bf16) ----
#pragma unroll
    for (int c = 0; c < 4; ++c)
#pragma unroll
        for (int v = 0; v < 8; ++v) {
            int sIdx = q0 + v + 8 * half;
            int feat = h * HDIM + c * 16 + m16;
            float val = o[c][v] / lrow[v];
            attn[(size_t)(b * SEQ + sIdx) * DMODEL + feat] = f2bf(val);
        }
}

// ---------------------------------------------------------------------------
// Host launcher.  Workspace layout (bytes):
//   [0,        8388608)  x_bf      4096x1024 bf16
//   [8388608, 14680064)  w1_bf     3072x1024 bf16
//   [14680064,16777216)  w2_bf     1024x1024 bf16
//   [16777216,41943040)  qkv_bf    4096x3072 bf16
//   [41943040,50331648)  attn_bf   4096x1024 bf16
// ---------------------------------------------------------------------------
extern "C" void kernel_launch(void* const* d_in, const int* in_sizes, int n_in,
                              void* d_out, int out_size, void* d_ws, size_t ws_size,
                              hipStream_t stream) {
    (void)in_sizes; (void)n_in; (void)out_size; (void)ws_size;
    const float* x  = (const float*)d_in[0];
    const float* w1 = (const float*)d_in[1];
    const float* b1 = (const float*)d_in[2];
    const float* w2 = (const float*)d_in[3];
    const float* b2 = (const float*)d_in[4];
    float* out = (float*)d_out;

    char* ws = (char*)d_ws;
    u16* xbf    = (u16*)(ws);
    u16* w1bf   = (u16*)(ws + 8388608);
    u16* w2bf   = (u16*)(ws + 14680064);
    u16* qkvbf  = (u16*)(ws + 16777216);
    u16* attnbf = (u16*)(ws + 41943040);

    // 1) bf16 conversions
    cvt_f32_bf16_x4<<<(NTOK * DMODEL / 4 + 255) / 256, 256, 0, stream>>>(x,  xbf,  NTOK * DMODEL / 4);
    cvt_f32_bf16_x4<<<(3 * DMODEL * DMODEL / 4 + 255) / 256, 256, 0, stream>>>(w1, w1bf, 3 * DMODEL * DMODEL / 4);
    cvt_f32_bf16_x4<<<(DMODEL * DMODEL / 4 + 255) / 256, 256, 0, stream>>>(w2, w2bf, DMODEL * DMODEL / 4);

    // 2) fused QKV projection: [4096,1024] x [3072,1024]^T -> bf16 [4096,3072]
    gemm_bf16_nt<false><<<dim3(3 * DMODEL / 128, NTOK / 128), 128, 0, stream>>>(
        xbf, w1bf, b1, qkvbf, 3 * DMODEL, DMODEL);

    // 3) flash attention over the local(+past) window
    flash_attn<<<dim3(2 * NHEAD, SEQ / 64), 128, 0, stream>>>(qkvbf, attnbf);

    // 4) output projection: [4096,1024] x [1024,1024]^T -> f32 d_out
    gemm_bf16_nt<true><<<dim3(DMODEL / 128, NTOK / 128), 128, 0, stream>>>(
        attnbf, w2bf, b2, out, DMODEL, DMODEL);
}
